// CausalSelfAttention_23184233464066
// MI455X (gfx1250) — matrix operand order repaired and checked by reading, not compiled, not run on hardware
//
#include <hip/hip_runtime.h>
#include <hip/hip_bf16.h>

typedef __attribute__((ext_vector_type(16))) _Float16 v16h;
typedef __attribute__((ext_vector_type(8)))  _Float16 v8h;
typedef __attribute__((ext_vector_type(8)))  float    v8f;
typedef __attribute__((ext_vector_type(4)))  unsigned int u32x4;
typedef __attribute__((ext_vector_type(8)))  int      i32x8;
typedef __attribute__((ext_vector_type(4)))  int      i32x4;

#define BATCH 4
#define SEQ   2048
#define EMBD  1024
#define NH    16
#define HD    64
#define QKV_N 3072   // 3*EMBD

#if defined(__has_builtin)
#if __has_builtin(__builtin_amdgcn_tensor_load_to_lds) && \
    __has_builtin(__builtin_amdgcn_s_wait_tensorcnt)
#define HAVE_TDM 1
#endif
#endif
#ifndef HAVE_TDM
#define HAVE_TDM 0
#endif

// ---------------------------------------------------------------------------
// WMMA fragment helpers (wave32; A/B use this code's contiguous K order)
// A 16x32 f16: lane<16 holds row m=lane, 16 contiguous K from 2*c0, c0=0;
//              lane>=16 same rows, c0=8 (K 16..31).
// B 32x16 f16 (as B^T rows): lane holds col n=lane%16, 16 contiguous K values
//              starting at (lane<16 ? 0 : 16).
// C/D 16x16 f32: vgpr i holds (m = i + 8*(lane>=16), n = lane%16).
// ---------------------------------------------------------------------------
__device__ __forceinline__ v16h load_a_frag(const _Float16* row, int c0) {
  v8h lo = *(const v8h*)(row + 2 * c0);
  v8h hi = *(const v8h*)(row + 2 * c0 + 8);
  return __builtin_shufflevector(lo, hi, 0,1,2,3,4,5,6,7,8,9,10,11,12,13,14,15);
}
__device__ __forceinline__ v16h load_b_frag(const _Float16* p) {
  v8h lo = *(const v8h*)(p);
  v8h hi = *(const v8h*)(p + 8);
  return __builtin_shufflevector(lo, hi, 0,1,2,3,4,5,6,7,8,9,10,11,12,13,14,15);
}
__device__ __forceinline__ v8f wmma_f16(v16h a, v16h b, v8f c) {
  return __builtin_amdgcn_wmma_f32_16x16x32_f16(false, a, false, b, (short)0, c,
                                                false, false);
}

// ---------------------------------------------------------------------------
// TDM: 2D tile (tile_x elems of data_size=2B per row, tile_y rows) from global
// to LDS, with LDS row padding: after every 32 DWORDs (64 halfs) insert
// 4 DWORDs (8 halfs) -> LDS pitch of 72 halfs, matching Kt[64][72].
// D# packing per CDNA5 ISA ch.8 (group0 128b, group1 256b).
// ---------------------------------------------------------------------------
#if HAVE_TDM
__device__ __forceinline__ void tdm_load_tile_2d(
    unsigned lds_addr, const _Float16* gptr, unsigned tile_x, unsigned tile_y,
    unsigned row_stride_elems, unsigned tensor_rows) {
  const unsigned long long ga = (unsigned long long)(size_t)gptr;
  u32x4 g0;
  g0[0] = 1u;                                        // count=1, user D#
  g0[1] = lds_addr;                                  // lds byte address
  g0[2] = (unsigned)(ga & 0xffffffffu);              // global_addr[31:0]
  g0[3] = (unsigned)((ga >> 32) & 0x01ffffffu)       // global_addr[56:32]
          | (2u << 30);                              // type=2 ("image")
  i32x8 g1;
  // [17:16]=data_size 2B, [20]=pad_enable, [24:22]=pad_interval(4 -> 32dw),
  // [31:25]=pad_amount(3 -> 4dw); workgroup_mask=0 (not in cluster)
  g1[0] = (1 << 16) | (1 << 20) | (4 << 22) | (3 << 25);
  g1[1] = (int)((tile_x & 0xffffu) << 16);           // tensor_dim0[15:0]
  g1[2] = (int)(((tile_x >> 16) & 0xffffu)           // tensor_dim0[31:16]
                | ((tensor_rows & 0xffffu) << 16));  // tensor_dim1[15:0]
  g1[3] = (int)(((tensor_rows >> 16) & 0xffffu)      // tensor_dim1[31:16]
                | ((tile_x & 0xffffu) << 16));       // tile_dim0
  g1[4] = (int)(tile_y & 0xffffu);                   // tile_dim1 (tile_dim2=0)
  g1[5] = (int)row_stride_elems;                     // tensor_dim0_stride[31:0]
  g1[6] = 0;                                         // stride hi / dim1_stride
  g1[7] = 0;
  const i32x4 gz = {0, 0, 0, 0};
#if __clang_major__ >= 23
  const i32x8 gz8 = {0, 0, 0, 0, 0, 0, 0, 0};
  __builtin_amdgcn_tensor_load_to_lds(g0, g1, gz, gz, gz8, 0);
#else
  __builtin_amdgcn_tensor_load_to_lds(g0, g1, gz, gz, 0);
#endif
}
#endif

// ---------------------------------------------------------------------------
// Tiled WMMA GEMM with bias:  Out[M,N] = A[M,K] * B[K,N] + bias[N]
// Block: 64x64 tile, BK=32, 128 threads = 4 waves, each wave a 32x32 quadrant.
// ---------------------------------------------------------------------------
template <typename TA, typename TO>
__global__ __launch_bounds__(128)
void wmma_gemm_bias(const TA* __restrict__ A, const float* __restrict__ Bw,
                    const float* __restrict__ bias, TO* __restrict__ Out,
                    int M, int N, int K) {
  __shared__ _Float16 As[64][40];   // [m][k], padded pitch (80B, 16B-aligned)
  __shared__ _Float16 Bt[64][40];   // [n][k]

  const int tid  = threadIdx.x;
  const int wave = tid >> 5;
  const int lane = tid & 31;
  const int hl   = lane >> 4;
  const int nl   = lane & 15;
  const int m0   = blockIdx.y * 64;
  const int n0   = blockIdx.x * 64;
  const int mq   = (wave >> 1) * 32;
  const int nq   = (wave & 1) * 32;

  const v8f zero8 = {0.f,0.f,0.f,0.f,0.f,0.f,0.f,0.f};
  v8f acc[2][2] = {{zero8, zero8}, {zero8, zero8}};

  const int lrow = tid >> 1;          // 0..63
  const int lseg = (tid & 1) * 16;    // 0 or 16

  for (int k0 = 0; k0 < K; k0 += 32) {
    __syncthreads();
    const TA* ag = A + (size_t)(m0 + lrow) * K + k0 + lseg;
#pragma unroll
    for (int j = 0; j < 16; ++j) As[lrow][lseg + j] = (_Float16)ag[j];
    const float* bg = Bw + (size_t)(k0 + lseg) * N + n0 + lrow;
#pragma unroll
    for (int j = 0; j < 16; ++j) Bt[lrow][lseg + j] = (_Float16)bg[(size_t)j * N];
    if (k0 + 32 < K) __builtin_prefetch(ag + 32, 0, 0);  // global_prefetch
    __syncthreads();

    v16h af[2], bf[2];
#pragma unroll
    for (int mt = 0; mt < 2; ++mt)
      af[mt] = load_a_frag(&As[mq + mt * 16 + nl][0], hl * 8);
#pragma unroll
    for (int nt = 0; nt < 2; ++nt)
      bf[nt] = load_b_frag(&Bt[nq + nt * 16 + nl][hl * 16]);
#pragma unroll
    for (int mt = 0; mt < 2; ++mt)
#pragma unroll
      for (int nt = 0; nt < 2; ++nt)
        acc[mt][nt] = wmma_f16(af[mt], bf[nt], acc[mt][nt]);
  }

#pragma unroll
  for (int mt = 0; mt < 2; ++mt)
#pragma unroll
    for (int nt = 0; nt < 2; ++nt) {
      const int col = n0 + nq + nt * 16 + nl;
      const float bv = bias[col];
#pragma unroll
      for (int i = 0; i < 8; ++i) {
        const int row = m0 + mq + mt * 16 + hl * 8 + i;
        Out[(size_t)row * N + col] = (TO)(acc[mt][nt][i] + bv);
      }
    }
}

// ---------------------------------------------------------------------------
// Flash attention: one block per (query-block of 64, b*H+h).
// 4 waves; wave w owns query rows [qb*64 + 16w, +16). Streams 64-key tiles
// with online softmax. K tile is DMA'd by the Tensor Data Mover (wave 0)
// while all threads stage V transposed; Q*K^T and P*V use WMMA f16.
// ---------------------------------------------------------------------------
__global__ __launch_bounds__(128)
void flash_attn_kernel(const _Float16* __restrict__ qkv,  // [B*T, 3C] f16
                       _Float16* __restrict__ Yo) {       // [B*T, C]  f16
  __shared__ _Float16 Kt[64][72];      // [key][d]   (== B^T for Q*K^T)
  __shared__ _Float16 Vt[64][72];      // [d][key]   (== B^T for P*V)
  __shared__ _Float16 Pl[4][16][72];   // per-wave P staging

  const int tid  = threadIdx.x;
  const int wave = tid >> 5;
  const int lane = tid & 31;
  const int hl   = lane >> 4;
  const int nl   = lane & 15;
  const int qb   = blockIdx.x;
  const int b    = blockIdx.y >> 4;
  const int h    = blockIdx.y & 15;

  const int t0w = qb * 64 + wave * 16;      // first query row of this wave
  const size_t rp = QKV_N;                  // row pitch in halves

  // Q fragments for this wave's 16 rows, kept in registers for whole kernel
  const _Float16* qrow = qkv + ((size_t)(b * SEQ) + t0w + nl) * rp + h * HD;
  const v16h aQ0 = load_a_frag(qrow, hl * 8);        // contraction d = 0..31
  const v16h aQ1 = load_a_frag(qrow + 32, hl * 8);   // contraction d = 32..63

  const v8f zero8 = {0.f,0.f,0.f,0.f,0.f,0.f,0.f,0.f};
  v8f yacc[4] = {zero8, zero8, zero8, zero8};
  float mrun[8], lrun[8];
#pragma unroll
  for (int i = 0; i < 8; ++i) { mrun[i] = -3.0e38f; lrun[i] = 0.f; }

  const int lrow = tid >> 1;           // 0..63 (key row for loaders)
  const int lseg = (tid & 1) * 32;     // 0 or 32 (d segment)
  const _Float16* kbase = qkv + (size_t)(b * SEQ) * rp + EMBD + h * HD;
  const _Float16* vbase = qkv + (size_t)(b * SEQ) * rp + 2 * EMBD + h * HD;
#if HAVE_TDM
  const unsigned kt_lds = (unsigned)(size_t)(void*)&Kt[0][0];
#endif

  for (int kb = 0; kb <= qb; ++kb) {
    __syncthreads();
#if HAVE_TDM
    // Tensor Data Mover: 64x64-half K tile, global row stride 3072 halves,
    // LDS pitch padded to 72 halves by TDM pad fields. Wave 0 issues; TDM
    // ignores EXEC and signals TENSORcnt once on completion.
    if (wave == 0)
      tdm_load_tile_2d(kt_lds, kbase + (size_t)(kb * 64) * rp,
                       /*tile_x=*/HD, /*tile_y=*/64,
                       /*row_stride=*/QKV_N, /*tensor_rows=*/SEQ);
#else
    const _Float16* kg = kbase + (size_t)(kb * 64 + lrow) * rp + lseg;
#pragma unroll
    for (int j = 0; j < 32; ++j) Kt[lrow][lseg + j] = kg[j];
#endif
    // All 128 threads stage V transposed (overlaps with the TDM DMA)
    const _Float16* vg = vbase + (size_t)(kb * 64 + lrow) * rp + lseg;
#pragma unroll
    for (int j = 0; j < 32; ++j) Vt[lseg + j][lrow] = vg[j];
#if HAVE_TDM
    if (wave == 0) __builtin_amdgcn_s_wait_tensorcnt(0);
#endif
    __syncthreads();

    // --- S = Q * K^T (16 rows x 64 keys = 4 tiles x 2 WMMAs) ---
    v8f s[4] = {zero8, zero8, zero8, zero8};
#pragma unroll
    for (int nt = 0; nt < 4; ++nt) {
      v16h bk0 = load_b_frag(&Kt[nt * 16 + nl][hl * 16]);
      v16h bk1 = load_b_frag(&Kt[nt * 16 + nl][32 + hl * 16]);
      s[nt] = wmma_f16(aQ0, bk0, s[nt]);
      s[nt] = wmma_f16(aQ1, bk1, s[nt]);
    }

    // scale + causal mask (diagonal block only; select, no divergence)
    const bool diag = (kb == qb);
#pragma unroll
    for (int nt = 0; nt < 4; ++nt)
#pragma unroll
      for (int i = 0; i < 8; ++i) {
        float v = s[nt][i] * 0.125f;           // 1/sqrt(64)
        if (diag) {
          const int qi = t0w + hl * 8 + i;
          const int kj = kb * 64 + nt * 16 + nl;
          v = (kj > qi) ? -3.0e38f : v;
        }
        s[nt][i] = v;
      }

    // --- online softmax: row max / rescale / exp / row sum ---
    float alpha[8];
#pragma unroll
    for (int i = 0; i < 8; ++i) {
      float v = fmaxf(fmaxf(s[0][i], s[1][i]), fmaxf(s[2][i], s[3][i]));
      v = fmaxf(v, __shfl_xor(v, 1, 32));
      v = fmaxf(v, __shfl_xor(v, 2, 32));
      v = fmaxf(v, __shfl_xor(v, 4, 32));
      v = fmaxf(v, __shfl_xor(v, 8, 32));
      const float mnew = fmaxf(mrun[i], v);
      alpha[i] = __expf(mrun[i] - mnew);
      mrun[i] = mnew;
    }
#pragma unroll
    for (int nt = 0; nt < 4; ++nt)
#pragma unroll
      for (int i = 0; i < 8; ++i) s[nt][i] = __expf(s[nt][i] - mrun[i]);
#pragma unroll
    for (int i = 0; i < 8; ++i) {
      float rs = (s[0][i] + s[1][i]) + (s[2][i] + s[3][i]);
      rs += __shfl_xor(rs, 1, 32);
      rs += __shfl_xor(rs, 2, 32);
      rs += __shfl_xor(rs, 4, 32);
      rs += __shfl_xor(rs, 8, 32);
      lrun[i] = lrun[i] * alpha[i] + rs;
    }
#pragma unroll
    for (int dt = 0; dt < 4; ++dt)
#pragma unroll
      for (int i = 0; i < 8; ++i) yacc[dt][i] *= alpha[i];

    // --- P (D-layout regs) -> LDS -> A-layout fragments ---
#pragma unroll
    for (int nt = 0; nt < 4; ++nt)
#pragma unroll
      for (int i = 0; i < 8; ++i)
        Pl[wave][hl * 8 + i][nt * 16 + nl] = (_Float16)s[nt][i];
    __syncthreads();
    const v16h aP0 = load_a_frag(&Pl[wave][nl][0], hl * 8);   // keys 0..31
    const v16h aP1 = load_a_frag(&Pl[wave][nl][32], hl * 8);  // keys 32..63

    // --- Y += P * V ---
#pragma unroll
    for (int dt = 0; dt < 4; ++dt) {
      v16h bv0 = load_b_frag(&Vt[dt * 16 + nl][hl * 16]);
      v16h bv1 = load_b_frag(&Vt[dt * 16 + nl][32 + hl * 16]);
      yacc[dt] = wmma_f16(aP0, bv0, yacc[dt]);
      yacc[dt] = wmma_f16(aP1, bv1, yacc[dt]);
    }
  }

  // epilogue: normalize and store f16 y in [B,T,C] layout
  float inv[8];
#pragma unroll
  for (int i = 0; i < 8; ++i) inv[i] = 1.0f / lrun[i];
#pragma unroll
  for (int dt = 0; dt < 4; ++dt)
#pragma unroll
    for (int i = 0; i < 8; ++i) {
      const int t = t0w + hl * 8 + i;
      Yo[((size_t)(b * SEQ) + t) * EMBD + h * HD + dt * 16 + nl] =
          (_Float16)(yacc[dt][i] * inv[i]);
    }
}

// ---------------------------------------------------------------------------
// Launch: qkv GEMM -> flash attention -> output projection GEMM
// ---------------------------------------------------------------------------
extern "C" void kernel_launch(void* const* d_in, const int* in_sizes, int n_in,
                              void* d_out, int out_size, void* d_ws, size_t ws_size,
                              hipStream_t stream) {
  const float* x      = (const float*)d_in[0];   // [B,T,C]
  const float* W_attn = (const float*)d_in[1];   // [C,3C]
  const float* b_attn = (const float*)d_in[2];   // [3C]
  const float* W_proj = (const float*)d_in[3];   // [C,C]
  const float* b_proj = (const float*)d_in[4];   // [C]
  float* out = (float*)d_out;                    // [B,T,C] f32

  const int M = BATCH * SEQ;   // 8192
  _Float16* qkv16 = (_Float16*)d_ws;                         // 8192*3072 f16
  _Float16* y16   = qkv16 + (size_t)M * QKV_N;               // 8192*1024 f16

  // 1) qkv = x @ W_attn + b_attn  -> f16
  wmma_gemm_bias<float, _Float16>
      <<<dim3(QKV_N / 64, M / 64), 128, 0, stream>>>(
          x, W_attn, b_attn, qkv16, M, QKV_N, EMBD);

  // 2) flash attention over [B,H] x query blocks of 64
  flash_attn_kernel<<<dim3(SEQ / 64, BATCH * NH), 128, 0, stream>>>(qkv16, y16);

  // 3) out = y @ W_proj + b_proj  -> f32
  wmma_gemm_bias<_Float16, float>
      <<<dim3(EMBD / 64, M / 64), 128, 0, stream>>>(
          y16, W_proj, b_proj, out, M, EMBD, EMBD);
}
